// EdgePooling_20658792694320
// MI455X (gfx1250) — matrix operand order repaired and checked by reading, not compiled, not run on hardware
//
#include <hip/hip_runtime.h>
#include <hip/hip_bf16.h>

// ---------------------------------------------------------------------------
// EdgePooling for MI455X (gfx1250).
//   inputs : x[N,C] f32, edge_index[2,E] i32, batch[N] i32, lin_w[2C] f32, lin_b f32
//   outputs (concat, as float): new_x[N,C] | new_edge[2E] | cluster[N] | new_batch[N] | num_clusters
// ---------------------------------------------------------------------------

#define C_DIM 256
#define MAXN  20480          // LDS avail[] capacity (N = 20000)
#define RADIX 256
#define ITEMS_PER_BLOCK 1024 // radix hist/scatter tile
#define MATCH_BATCH 2048

typedef __attribute__((ext_vector_type(16))) _Float16 v16h;
typedef __attribute__((ext_vector_type(8)))  float    v8f;

// --------------------------- node score GEMV via WMMA ----------------------
// s_lo[n] = dot(x[n,:], w[0:C]) ; s_hi[n] = dot(x[n,:], w[C:2C])
// One wave computes a 16-node tile: D(16x16) = A(16x32 f16) x B(32x16 f16),
// accumulated over 8 K-steps (K=256). B column 0 = w_lo chunk, column 1 = w_hi.
// All fragment loads are unconditional float4 (global_load_b128); row overrun
// is handled by clamping (overrun D rows are simply never stored), and the
// B column selection is a per-lane base pointer + mask multiply (no EXEC churn).
__global__ __launch_bounds__(256)
void k_node_scores(const float* __restrict__ x, const float* __restrict__ w,
                   float* __restrict__ sLo, float* __restrict__ sHi, int nNodes)
{
    const int waveId = (blockIdx.x * blockDim.x + threadIdx.x) >> 5;
    const int lane   = threadIdx.x & 31;
    const int tileBase = waveId * 16;
    if (tileBase >= nNodes) return;            // wave-uniform exit

    const int col = lane & 15;                 // N column held by this lane (B/D)
    const int grp = lane >> 4;                 // lane half
    // A row for this lane; clamp so padded rows read row (nNodes-1) harmlessly.
    const int rowA = min(tileBase + (lane & 15), nNodes - 1);
    const float* __restrict__ xr = x + (size_t)rowA * C_DIM;

    // Per-lane B source: column 0 -> w_lo, column 1 -> w_hi, others masked to 0.
    const float* __restrict__ wp = w + ((col == 1) ? C_DIM : 0);
    const float msk = (col < 2) ? 1.0f : 0.0f;

    v8f acc = {};
    #pragma unroll
    for (int kbase = 0; kbase < C_DIM; kbase += 32) {
        // ---- A fragment: lanes 0-15 hold K {0..7,16..23}, lanes 16-31 K {8..15,24..31}
        const float4 a0 = *(const float4*)(xr + kbase + grp * 8);
        const float4 a1 = *(const float4*)(xr + kbase + grp * 8 + 4);
        const float4 a2 = *(const float4*)(xr + kbase + 16 + grp * 8);
        const float4 a3 = *(const float4*)(xr + kbase + 16 + grp * 8 + 4);
        v16h a;
        a[0]  = (_Float16)a0.x; a[1]  = (_Float16)a0.y;
        a[2]  = (_Float16)a0.z; a[3]  = (_Float16)a0.w;
        a[4]  = (_Float16)a1.x; a[5]  = (_Float16)a1.y;
        a[6]  = (_Float16)a1.z; a[7]  = (_Float16)a1.w;
        a[8]  = (_Float16)a2.x; a[9]  = (_Float16)a2.y;
        a[10] = (_Float16)a2.z; a[11] = (_Float16)a2.w;
        a[12] = (_Float16)a3.x; a[13] = (_Float16)a3.y;
        a[14] = (_Float16)a3.z; a[15] = (_Float16)a3.w;

        // ---- B fragment: lane = N column; lanes 0-15 hold K {0..7,16..23}, lanes 16-31 K {8..15,24..31}
        const int K0 = kbase + grp * 8;
        const float4 w0 = *(const float4*)(wp + K0);
        const float4 w1 = *(const float4*)(wp + K0 + 4);
        const float4 w2 = *(const float4*)(wp + K0 + 16);
        const float4 w3 = *(const float4*)(wp + K0 + 20);
        v16h b;
        b[0]  = (_Float16)(w0.x * msk); b[1]  = (_Float16)(w0.y * msk);
        b[2]  = (_Float16)(w0.z * msk); b[3]  = (_Float16)(w0.w * msk);
        b[4]  = (_Float16)(w1.x * msk); b[5]  = (_Float16)(w1.y * msk);
        b[6]  = (_Float16)(w1.z * msk); b[7]  = (_Float16)(w1.w * msk);
        b[8]  = (_Float16)(w2.x * msk); b[9]  = (_Float16)(w2.y * msk);
        b[10] = (_Float16)(w2.z * msk); b[11] = (_Float16)(w2.w * msk);
        b[12] = (_Float16)(w3.x * msk); b[13] = (_Float16)(w3.y * msk);
        b[14] = (_Float16)(w3.z * msk); b[15] = (_Float16)(w3.w * msk);

        acc = __builtin_amdgcn_wmma_f32_16x16x32_f16(
                  false, a, false, b, (short)0, acc, false, false);
    }

    // D layout: VGPR r -> row (grp ? 8+r : r), N = lane&15. Columns 0/1 are s_lo/s_hi.
    if (col < 2) {
        float* __restrict__ outp = (col == 0) ? sLo : sHi;
        #pragma unroll
        for (int r = 0; r < 8; ++r) {
            const int node = tileBase + grp * 8 + r;
            if (node < nNodes) outp[node] = acc[r];
        }
    }
}

// --------------------------- edge scores + radix keys ----------------------
__global__ void k_edge_scores(const float* __restrict__ sLo, const float* __restrict__ sHi,
                              const int* __restrict__ src, const int* __restrict__ dst,
                              const float* __restrict__ biasPtr, int nE,
                              float* __restrict__ e, unsigned* __restrict__ keys,
                              unsigned* __restrict__ pay)
{
    const int i = blockIdx.x * blockDim.x + threadIdx.x;
    if (i >= nE) return;
    const float bias = biasPtr[0];
    float v = tanhf(sLo[src[i]] + sHi[dst[i]] + bias) + 0.5f;
    e[i] = v;
    unsigned u = __float_as_uint(v);
    u ^= (u >> 31) ? 0xFFFFFFFFu : 0x80000000u;  // ascending-sortable
    keys[i] = ~u;                                 // invert -> descending score
    pay[i]  = (unsigned)i;
}

// --------------------------- LSD radix sort --------------------------------
__global__ void k_hist(const unsigned* __restrict__ keys, int n, int shift,
                       unsigned* __restrict__ hist, int nBlocks)
{
    __shared__ unsigned h[RADIX];
    for (int i = threadIdx.x; i < RADIX; i += blockDim.x) h[i] = 0u;
    __syncthreads();
    const int base = blockIdx.x * ITEMS_PER_BLOCK;
    for (int t = threadIdx.x; t < ITEMS_PER_BLOCK; t += blockDim.x) {
        const int i = base + t;
        if (i + 4096 < n) __builtin_prefetch(&keys[i + 4096], 0, 0); // global_prefetch_b8
        if (i < n) atomicAdd(&h[(keys[i] >> shift) & 255u], 1u);
    }
    __syncthreads();
    for (int d = threadIdx.x; d < RADIX; d += blockDim.x)
        hist[(size_t)d * nBlocks + blockIdx.x] = h[d];   // bin-major for global scan
}

__global__ void k_scan(unsigned* __restrict__ hist, int total)
{
    __shared__ unsigned sums[256];
    const int chunk = (total + 255) / 256;
    const int beg = threadIdx.x * chunk;
    const int end = min(beg + chunk, total);
    unsigned s = 0;
    for (int i = beg; i < end; ++i) s += hist[i];
    sums[threadIdx.x] = s;
    __syncthreads();
    if (threadIdx.x == 0) {
        unsigned run = 0;
        for (int i = 0; i < 256; ++i) { unsigned v = sums[i]; sums[i] = run; run += v; }
    }
    __syncthreads();
    unsigned run = sums[threadIdx.x];
    for (int i = beg; i < end; ++i) { unsigned v = hist[i]; hist[i] = run; run += v; }
}

__global__ void k_scatter(const unsigned* __restrict__ keys, const unsigned* __restrict__ pay,
                          int n, int shift, unsigned* __restrict__ hist, int nBlocks,
                          unsigned* __restrict__ okeys, unsigned* __restrict__ opay)
{
    const int base = blockIdx.x * ITEMS_PER_BLOCK;
    for (int t = threadIdx.x; t < ITEMS_PER_BLOCK; t += blockDim.x) {
        const int i = base + t;
        if (i < n) {
            const unsigned k = keys[i];
            const unsigned d = (k >> shift) & 255u;
            const unsigned pos = atomicAdd(&hist[(size_t)d * nBlocks + blockIdx.x], 1u);
            okeys[pos] = k;
            opay[pos]  = pay[i];
        }
    }
}

// --------------------------- sequential greedy matching --------------------
// Single workgroup. avail[] lives in LDS; 256 threads stage sorted-edge batches
// into LDS, lane 0 resolves serially (the algorithm is order-dependent).
__global__ __launch_bounds__(256)
void k_match(const unsigned* __restrict__ pay, const float* __restrict__ e,
             const int* __restrict__ src, const int* __restrict__ dst,
             int nE, int nNodes, int* __restrict__ cluster,
             float* __restrict__ cscore, int* __restrict__ scalars)
{
    __shared__ unsigned char avail[MAXN];
    __shared__ int   bs[MATCH_BATCH];
    __shared__ int   bd[MATCH_BATCH];
    __shared__ float be[MATCH_BATCH];
    __shared__ int   s_cid;

    for (int i = threadIdx.x; i < nNodes; i += blockDim.x) {
        avail[i]   = 1;
        cscore[i]  = 1.0f;
        cluster[i] = 0;
    }
    if (threadIdx.x == 0) s_cid = 0;
    __syncthreads();

    for (int base = 0; base < nE; base += MATCH_BATCH) {
        const int cnt = min(MATCH_BATCH, nE - base);
        for (int t = threadIdx.x; t < cnt; t += blockDim.x) {
            const int p = (int)pay[base + t];
            bs[t] = src[p];
            bd[t] = dst[p];
            be[t] = e[p];
        }
        __syncthreads();
        if (threadIdx.x == 0) {
            int cid = s_cid;
            for (int t = 0; t < cnt; ++t) {
                const int s = bs[t], d = bd[t];
                if (avail[s] & avail[d]) {
                    avail[s] = 0; avail[d] = 0;
                    cluster[s] = cid; cluster[d] = cid;
                    cscore[cid] = be[t];
                    ++cid;
                }
            }
            s_cid = cid;
        }
        __syncthreads();
    }

    if (threadIdx.x == 0) {
        int cid = s_cid;
        const int nMatched = cid;
        for (int n = 0; n < nNodes; ++n)
            if (avail[n]) { cluster[n] = cid; cscore[cid] = 1.0f; ++cid; }
        scalars[0] = nMatched;
        scalars[1] = cid;          // num_clusters
    }
}

// --------------------------- pooling & outputs -----------------------------
__global__ void k_zero(float* __restrict__ p, long long n)
{
    long long i = (long long)blockIdx.x * blockDim.x + threadIdx.x;
    long long stride = (long long)gridDim.x * blockDim.x;
    for (; i < n; i += stride) p[i] = 0.0f;
}

__global__ __launch_bounds__(C_DIM)
void k_pool(const float* __restrict__ x, const int* __restrict__ cluster,
            float* __restrict__ newx, int nNodes)
{
    const int n = blockIdx.x;
    if (n >= nNodes) return;
    const int k = cluster[n];
    const int c = threadIdx.x;
    atomicAdd(&newx[(size_t)k * C_DIM + c], x[(size_t)n * C_DIM + c]);
}

__global__ __launch_bounds__(C_DIM)
void k_scale(float* __restrict__ newx, const float* __restrict__ cscore, int nNodes)
{
    const int n = blockIdx.x;
    if (n >= nNodes) return;
    newx[(size_t)n * C_DIM + threadIdx.x] *= cscore[n];
}

__global__ void k_remap(const int* __restrict__ ei, const int* __restrict__ cluster,
                        float* __restrict__ out, int n2E)
{
    const int i = blockIdx.x * blockDim.x + threadIdx.x;
    if (i < n2E) out[i] = (float)cluster[ei[i]];
}

__global__ void k_batch_scatter(const int* __restrict__ cluster, const int* __restrict__ batch,
                                float* __restrict__ outBatch, int nNodes)
{
    const int i = blockIdx.x * blockDim.x + threadIdx.x;
    if (i < nNodes) outBatch[cluster[i]] = (float)batch[i];
}

__global__ void k_tail(const int* __restrict__ cluster, const int* __restrict__ scalars,
                       float* __restrict__ outCluster, float* __restrict__ outNum, int nNodes)
{
    const int i = blockIdx.x * blockDim.x + threadIdx.x;
    if (i < nNodes) outCluster[i] = (float)cluster[i];
    if (i == 0) outNum[0] = (float)scalars[1];
}

// ---------------------------------------------------------------------------
extern "C" void kernel_launch(void* const* d_in, const int* in_sizes, int n_in,
                              void* d_out, int out_size, void* d_ws, size_t ws_size,
                              hipStream_t stream)
{
    const float* x     = (const float*)d_in[0];
    const int*   ei    = (const int*)  d_in[1];
    const int*   batch = (const int*)  d_in[2];
    const float* w     = (const float*)d_in[3];
    const float* bptr  = (const float*)d_in[4];

    const int N  = in_sizes[0] / C_DIM;   // 20000
    const int E  = in_sizes[1] / 2;       // 320000
    const int* src = ei;
    const int* dst = ei + E;

    // ---- workspace carve (256B aligned) ----
    char* ws = (char*)d_ws;
    size_t off = 0;
    auto carve = [&](size_t bytes) -> char* {
        char* p = ws + off;
        off = (off + bytes + 255) & ~(size_t)255;
        return p;
    };
    float*    sLo   = (float*)   carve((size_t)N * 4);
    float*    sHi   = (float*)   carve((size_t)N * 4);
    float*    eSc   = (float*)   carve((size_t)E * 4);
    unsigned* keysA = (unsigned*)carve((size_t)E * 4);
    unsigned* keysB = (unsigned*)carve((size_t)E * 4);
    unsigned* payA  = (unsigned*)carve((size_t)E * 4);
    unsigned* payB  = (unsigned*)carve((size_t)E * 4);
    const int nBlk  = (E + ITEMS_PER_BLOCK - 1) / ITEMS_PER_BLOCK;
    unsigned* hist  = (unsigned*)carve((size_t)RADIX * nBlk * 4);
    int*      clus  = (int*)     carve((size_t)N * 4);
    float*    csc   = (float*)   carve((size_t)N * 4);
    int*      scal  = (int*)     carve(2 * 4);

    // ---- output carve ----
    float* outX   = (float*)d_out;                   // N*C
    float* outE   = outX + (size_t)N * C_DIM;        // 2E
    float* outCl  = outE + (size_t)2 * E;            // N
    float* outB   = outCl + N;                       // N
    float* outNum = outB + N;                        // 1

    // 1) per-node score dots (WMMA GEMV)
    {
        const int waves  = (N + 15) / 16;
        const int blocks = (waves + 7) / 8;          // 8 waves per 256-thread block
        k_node_scores<<<blocks, 256, 0, stream>>>(x, w, sLo, sHi, N);
    }

    // 2) edge scores + sortable keys
    k_edge_scores<<<(E + 255) / 256, 256, 0, stream>>>(sLo, sHi, src, dst, bptr,
                                                       E, eSc, keysA, payA);

    // 3) 4-pass radix sort (descending by score); ends back in keysA/payA
    unsigned* ki = keysA; unsigned* pi = payA;
    unsigned* ko = keysB; unsigned* po = payB;
    for (int pass = 0; pass < 4; ++pass) {
        const int shift = pass * 8;
        k_hist   <<<nBlk, 256, 0, stream>>>(ki, E, shift, hist, nBlk);
        k_scan   <<<1,    256, 0, stream>>>(hist, RADIX * nBlk);
        k_scatter<<<nBlk, 256, 0, stream>>>(ki, pi, E, shift, hist, nBlk, ko, po);
        unsigned* t;
        t = ki; ki = ko; ko = t;
        t = pi; pi = po; po = t;
    }

    // 4) sequential greedy matching (single workgroup, LDS-resident avail[])
    k_match<<<1, 256, 0, stream>>>(pi, eSc, src, dst, E, N, clus, csc, scal);

    // 5) pooling: zero, scatter-add, scale by cluster score
    k_zero <<<512, 256, 0, stream>>>(outX, (long long)N * C_DIM);
    k_pool <<<N, C_DIM, 0, stream>>>(x, clus, outX, N);
    k_scale<<<N, C_DIM, 0, stream>>>(outX, csc, N);

    // 6) remapped edges, new batch, cluster ids, num_clusters
    k_remap<<<(2 * E + 255) / 256, 256, 0, stream>>>(ei, clus, outE, 2 * E);
    k_zero <<<64, 256, 0, stream>>>(outB, (long long)N);
    k_batch_scatter<<<(N + 255) / 256, 256, 0, stream>>>(clus, batch, outB, N);
    k_tail <<<(N + 255) / 256, 256, 0, stream>>>(clus, scal, outCl, outNum, N);
}
